// GraphAttention_30434138260138
// MI455X (gfx1250) — matrix operand order repaired and checked by reading, not compiled, not run on hardware
//
#include <hip/hip_runtime.h>

// ---- problem constants (match reference) ----
static constexpr int kB   = 8;
static constexpr int kC   = 128;
static constexpr int kN   = 2048;
static constexpr int kH   = 4;
static constexpr int kHD  = 32;
static constexpr int kOUT = 128;

typedef __attribute__((ext_vector_type(16))) _Float16 v16h;
typedef __attribute__((ext_vector_type(8)))  float    v8f;

// ---------------------------------------------------------------------------
// Fragment builders (wave32; this code's contiguous K order, same for A and B)
//   A 16x32 f16 : lane -> M=l16, VGPR i: K=g*16+2i (contiguous K pairs)
//   B 32x16 f16 : lane -> col=l16, VGPR i: K=g*16+2i (contiguous K pairs)
//   C/D 16x16 f32: VGPR r, lane -> (M = r+8g, N = l16)
// ---------------------------------------------------------------------------

__device__ __forceinline__ v16h load_a_f16(const _Float16* base, int lda, int g, int l16) {
  v16h a;
#pragma unroll
  for (int i = 0; i < 8; ++i) {
    int kb = g * 16 + i * 2;
    a[2 * i]     = base[l16 * lda + kb];
    a[2 * i + 1] = base[l16 * lda + kb + 1];
  }
  return a;
}

__device__ __forceinline__ v16h load_a_f32_kstride(const float* base, long kStride, int g, int l16) {
  v16h a;
#pragma unroll
  for (int i = 0; i < 8; ++i) {
    int kb = g * 16 + i * 2;
    a[2 * i]     = (_Float16)base[(long)kb * kStride + l16];
    a[2 * i + 1] = (_Float16)base[(long)(kb + 1) * kStride + l16];
  }
  return a;
}

__device__ __forceinline__ v16h load_b_f16(const _Float16* base, long colStride, int g, int l16) {
  v16h b;
  const _Float16* p = base + (long)l16 * colStride + g * 16;
#pragma unroll
  for (int i = 0; i < 8; ++i) {
    b[2 * i]     = p[2 * i];
    b[2 * i + 1] = p[2 * i + 1];
  }
  return b;
}

__device__ __forceinline__ v16h load_b_f32_kstride(const float* base, long kStride, int g, int l16) {
  v16h b;
  const float* p = base + l16 + (long)(g * 16) * kStride;
#pragma unroll
  for (int i = 0; i < 8; ++i) {
    b[2 * i]     = (_Float16)p[(long)(2 * i) * kStride];
    b[2 * i + 1] = (_Float16)p[(long)(2 * i + 1) * kStride];
  }
  return b;
}

// reductions across a 16-lane half (rows of a D fragment live in one half-wave)
__device__ __forceinline__ float half_red_max(float v) {
#pragma unroll
  for (int m = 1; m < 16; m <<= 1) v = fmaxf(v, __shfl_xor(v, m, 32));
  return v;
}
__device__ __forceinline__ float half_red_sum(float v) {
#pragma unroll
  for (int m = 1; m < 16; m <<= 1) v += __shfl_xor(v, m, 32);
  return v;
}

// ---------------------------------------------------------------------------
// Kernel 1: Q/K/V projection.  One wave -> one 16x16 tile of Q|K|V.
//   q,k stored f16 [b,h,n,32]; v stored transposed f16 [b,h,32,n]
// ---------------------------------------------------------------------------
__global__ void ga_qkv_kernel(const float* __restrict__ x,
                              const float* __restrict__ Wq,
                              const float* __restrict__ Wk,
                              const float* __restrict__ Wv,
                              _Float16* __restrict__ qws,
                              _Float16* __restrict__ kws,
                              _Float16* __restrict__ vtws) {
  const int lane = threadIdx.x & 31;
  const int g = lane >> 4, l16 = lane & 15;
  const int bt = blockIdx.x;            // b * (N/16) + row-tile
  const int b  = bt >> 7;               // N/16 == 128
  const int n0 = (bt & 127) * 16;
  const int j0 = blockIdx.y * 16;       // output column tile (0..127)
  const int which = blockIdx.z;         // 0=Q 1=K 2=V
  const float* W = (which == 0) ? Wq : ((which == 1) ? Wk : Wv);
  const int h   = j0 >> 5;              // head
  const int dl0 = j0 & 31;              // d offset within head

  v8f acc = {};
#pragma unroll
  for (int c0 = 0; c0 < kC; c0 += 32) {
    // A: x_t tile, element (M,K) = x[b, c0+K, n0+M]  (K stride = N)
    v16h a  = load_a_f32_kstride(x + (long)b * kC * kN + (long)c0 * kN + n0, kN, g, l16);
    // B: W tile, element (K,col) = W[h, c0+K, dl0+col] (K stride = HD)
    v16h bb = load_b_f32_kstride(W + (long)h * kC * kHD + (long)c0 * kHD + dl0, kHD, g, l16);
    acc = __builtin_amdgcn_wmma_f32_16x16x32_f16(false, a, false, bb, (short)0, acc, false, false);
  }

  const int  d     = dl0 + l16;
  const long nbase = (long)(b * kH + h) * kN;
#pragma unroll
  for (int r = 0; r < 8; ++r) {
    int n = n0 + r + 8 * g;
    _Float16 v = (_Float16)acc[r];
    if (which == 0)      qws[(nbase + n) * kHD + d] = v;
    else if (which == 1) kws[(nbase + n) * kHD + d] = v;
    else                 vtws[((long)(b * kH + h) * kHD + d) * kN + n] = v;
  }
}

// ---------------------------------------------------------------------------
// Kernel 2: attention.  One block = (b, 16-row tile); 4 waves = 4 heads.
//   pass 1 (per wave): online softmax stats (m, l) over 128 key tiles
//   pass 2 (per 32-key slab): p = exp(s-m)/l ; head-sum accumulated in LDS
//     via ds_add_f32; adj written to HBM exactly once (coalesced float4);
//     each wave accumulates its own P@V via WMMA.
// ---------------------------------------------------------------------------
__global__ void ga_attn_kernel(const _Float16* __restrict__ qws,
                               const _Float16* __restrict__ kws,
                               const _Float16* __restrict__ vtws,
                               _Float16* __restrict__ ocat,
                               float* __restrict__ adj) {
  __shared__ _Float16 stag[kH][16 * 32];   // per-wave D->A re-layout staging (4 KB)
  __shared__ float    adj_acc[16 * 32];    // head-summed prob tile (2 KB)

  const int tid  = threadIdx.x;
  const int lane = tid & 31;
  const int w    = tid >> 5;               // wave index == head
  const int g = lane >> 4, l16 = lane & 15;
  const int bt = blockIdx.x;               // b * (N/16) + row-tile
  const int b  = bt >> 7;
  const int n0 = (bt & 127) * 16;
  const int h  = w;
  const float scale = 0.17677669529663687f;   // 1/sqrt(32)

  const _Float16* qbase = qws  + ((long)(b * kH + h) * kN + n0) * kHD;
  const _Float16* kbase = kws  + (long)(b * kH + h) * kN * kHD;
  const _Float16* vbase = vtws + (long)(b * kH + h) * kHD * kN;

  v16h aq = load_a_f16(qbase, kHD, g, l16);   // Q rows, reused for all key tiles

  float m[8], l[8];
#pragma unroll
  for (int r = 0; r < 8; ++r) { m[r] = -1e30f; l[r] = 0.f; }

  // ---- pass 1: online row max / exp-sum (waves fully independent) ----
  for (int j = 0; j < kN / 16; ++j) {
    v8f z = {};
    v16h bk = load_b_f16(kbase + (long)j * 16 * kHD, kHD, g, l16); // (K=d, col=m)
    v8f s = __builtin_amdgcn_wmma_f32_16x16x32_f16(false, aq, false, bk, (short)0, z, false, false);
#pragma unroll
    for (int r = 0; r < 8; ++r) {
      float v  = s[r] * scale;
      float mn = fmaxf(m[r], half_red_max(v));
      float ts = half_red_sum(__expf(v - mn));
      l[r] = l[r] * __expf(m[r] - mn) + ts;
      m[r] = mn;
    }
  }
  float inv_l[8];
#pragma unroll
  for (int r = 0; r < 8; ++r) inv_l[r] = 1.0f / l[r];

  // ---- pass 2: probs -> LDS head-sum -> single adj store; P@V per head ----
  v8f co0 = {}, co1 = {};
  for (int j2 = 0; j2 < kN / 16; j2 += 2) {
    // zero the shared accumulator tile (16x32)
    for (int i = tid; i < 16 * 32; i += 128) adj_acc[i] = 0.f;
    __syncthreads();

#pragma unroll
    for (int js = 0; js < 2; ++js) {
      int j = j2 + js;
      v8f z = {};
      v16h bk = load_b_f16(kbase + (long)j * 16 * kHD, kHD, g, l16);
      v8f s = __builtin_amdgcn_wmma_f32_16x16x32_f16(false, aq, false, bk, (short)0, z, false, false);
#pragma unroll
      for (int r = 0; r < 8; ++r) {
        float p = __expf(s[r] * scale - m[r]) * inv_l[r];
        int idx = (r + 8 * g) * 32 + js * 16 + l16;
        stag[w][idx] = (_Float16)p;
        atomicAdd(&adj_acc[idx], p);      // ds_add_f32 (on-chip head sum)
      }
    }
    __syncthreads();

    // each wave: P@V for its head over this 32-key slab
    v16h ap  = load_a_f16(&stag[w][0], 32, g, l16);      // 16 rows x 32 keys
    // V^T: element (K=m_local, col=d) = vtws[(d)*N + j2*16 + K]  (K contiguous)
    v16h bv0 = load_b_f16(vbase + (long)j2 * 16,                 kN, g, l16);
    v16h bv1 = load_b_f16(vbase + (long)16 * kN + (long)j2 * 16, kN, g, l16);
    co0 = __builtin_amdgcn_wmma_f32_16x16x32_f16(false, ap, false, bv0, (short)0, co0, false, false);
    co1 = __builtin_amdgcn_wmma_f32_16x16x32_f16(false, ap, false, bv1, (short)0, co1, false, false);

    // cooperative coalesced write of the head-mean tile: 128 threads x float4
    {
      int row = tid >> 3;            // 0..15
      int c4  = (tid & 7) * 4;       // 0,4,...,28
      float4 v;
      v.x = 0.25f * adj_acc[row * 32 + c4 + 0];
      v.y = 0.25f * adj_acc[row * 32 + c4 + 1];
      v.z = 0.25f * adj_acc[row * 32 + c4 + 2];
      v.w = 0.25f * adj_acc[row * 32 + c4 + 3];
      *(float4*)&adj[((long)b * kN + n0 + row) * kN + j2 * 16 + c4] = v;
    }
    __syncthreads();   // protect adj_acc against next iteration's zeroing
  }

  // write concatenated-head output [b, n, h*32+d] f16
#pragma unroll
  for (int r = 0; r < 8; ++r) {
    int n = n0 + r + 8 * g;
    ocat[((long)b * kN + n) * kOUT + h * kHD + l16]      = (_Float16)co0[r];
    ocat[((long)b * kN + n) * kOUT + h * kHD + 16 + l16] = (_Float16)co1[r];
  }
}

// ---------------------------------------------------------------------------
// Kernel 3: out = out_cat @ W_o, stored transposed into d_out[b, o, n]
// ---------------------------------------------------------------------------
__global__ void ga_proj_kernel(const _Float16* __restrict__ ocat,
                               const float* __restrict__ Wo,
                               float* __restrict__ outT) {
  const int lane = threadIdx.x & 31;
  const int g = lane >> 4, l16 = lane & 15;
  const int bt = blockIdx.x;
  const int b  = bt >> 7;
  const int n0 = (bt & 127) * 16;
  const int o0 = blockIdx.y * 16;

  v8f acc = {};
#pragma unroll
  for (int k0 = 0; k0 < kOUT; k0 += 32) {
    v16h a  = load_a_f16(ocat + ((long)b * kN + n0) * kOUT + k0, kOUT, g, l16);
    v16h bb = load_b_f32_kstride(Wo + (long)k0 * kOUT + o0, kOUT, g, l16);
    acc = __builtin_amdgcn_wmma_f32_16x16x32_f16(false, a, false, bb, (short)0, acc, false, false);
  }
#pragma unroll
  for (int r = 0; r < 8; ++r) {
    int n = n0 + r + 8 * g;
    int o = o0 + l16;
    outT[((long)b * kOUT + o) * kN + n] = acc[r];
  }
}

// ---------------------------------------------------------------------------
// Launch
// ---------------------------------------------------------------------------
extern "C" void kernel_launch(void* const* d_in, const int* in_sizes, int n_in,
                              void* d_out, int out_size, void* d_ws, size_t ws_size,
                              hipStream_t stream) {
  const float* x  = (const float*)d_in[0];   // [B, C, N]
  const float* Wq = (const float*)d_in[1];   // [H, C, HD]
  const float* Wk = (const float*)d_in[2];
  const float* Wv = (const float*)d_in[3];
  const float* Wo = (const float*)d_in[4];   // [OUT, OUT]

  float* outT = (float*)d_out;                                 // [B, OUT, N]
  float* adj  = outT + (long)kB * kOUT * kN;                   // [B, N, N]

  // workspace: Q, K (4 MB each, f16 [b,h,n,32]), V^T (4 MB, f16 [b,h,32,n]),
  // out_cat (4 MB, f16 [b,n,128]) -> 16 MB total (L2-resident on MI455X)
  const long qkElems = (long)kB * kH * kN * kHD;               // 2,097,152 halves
  char* wsb = (char*)d_ws;
  _Float16* qws  = (_Float16*)(wsb);
  _Float16* kws  = (_Float16*)(wsb + 1 * qkElems * 2);
  _Float16* vtws = (_Float16*)(wsb + 2 * qkElems * 2);
  _Float16* ocat = (_Float16*)(wsb + 3 * qkElems * 2);

  // 1) Q/K/V projections: B*(N/16) x 8 col-tiles x {q,k,v}
  ga_qkv_kernel<<<dim3(kB * (kN / 16), kOUT / 16, 3), dim3(32), 0, stream>>>(
      x, Wq, Wk, Wv, qws, kws, vtws);

  // 2) attention: one block per (b, 16-row tile); 4 waves = 4 heads.
  //    adj is written exactly once, fully coalesced, no global atomics.
  ga_attn_kernel<<<dim3(kB * (kN / 16)), dim3(128), 0, stream>>>(
      qws, kws, vtws, ocat, adj);

  // 3) output projection (+transpose)
  ga_proj_kernel<<<dim3(kB * (kN / 16), kOUT / 16), dim3(32), 0, stream>>>(
      ocat, Wo, outT);
}
